// MSDeformableAttention_64046552318315
// MI455X (gfx1250) — compile-verified, matrix operand order checked by reading, not run on hardware
//
#include <hip/hip_runtime.h>
#include <hip/hip_bf16.h>

#define EMBED 256
#define KDIM  256
#define HEADS 8
#define LEVELS 4
#define POINTS 4
#define HEAD_DIM 32

typedef __attribute__((ext_vector_type(16))) _Float16 v16h;
typedef __attribute__((ext_vector_type(8)))  _Float16 v8h;
typedef __attribute__((ext_vector_type(8)))  float    v8f;
typedef __attribute__((ext_vector_type(4)))  unsigned u32x4;
typedef __attribute__((ext_vector_type(8)))  int      i32x8;
typedef __attribute__((ext_vector_type(4)))  int      i32x4;

union V16 { v16h v; v8h h[2]; };

// ---------------------------------------------------------------------------
// f32 -> f16 elementwise conversion (for the small weight matrices).
// ---------------------------------------------------------------------------
__global__ void f32_to_f16_kernel(const float* __restrict__ in,
                                  _Float16* __restrict__ out, int n)
{
    int i = blockIdx.x * blockDim.x + threadIdx.x;
    if (i < n) out[i] = (_Float16)in[i];
}

// ---------------------------------------------------------------------------
// WMMA GEMM:  out[M,N] = A[M,256] @ Wf16[N,256]^T + bias[N]
// Grid: (ceil(tilesM/8), N/64). Block = 256 threads = 8 waves.
// The 64x256 f16 W-strip (32 KB) is DMA'd into LDS by the Tensor Data Mover
// (one tensor_load_to_lds issued by wave 0, s_wait_tensorcnt, then barrier).
// Each wave computes a 16(M) x 64(N) slab: 1 A fragment feeds 4 v_wmma/K-step.
// N must be a multiple of 64; K fixed at 256.
// ---------------------------------------------------------------------------
__global__ __launch_bounds__(256) void wmma_gemm_bias_kernel(
    const float*    __restrict__ A,    // [M,K] f32
    const _Float16* __restrict__ Wf16, // [N,K] f16
    const float*    __restrict__ bias, // [N]
    float*          __restrict__ out,  // [M,N]
    int M, int N)
{
    __shared__ alignas(16) _Float16 smem[64 * KDIM];   // 32 KB f16 W-strip

    const int wave = threadIdx.x >> 5;
    const int lane = threadIdx.x & 31;
    const int nBase = blockIdx.y * 64;

    // ---- TDM: DMA W[nBase:nBase+64, 0:256] (f16, contiguous) -> LDS --------
    if (wave == 0) {
        unsigned long long ga = (unsigned long long)(Wf16 + (size_t)nBase * KDIM);
        unsigned lds_off = (unsigned)(size_t)(&smem[0]);   // LDS byte offset

        u32x4 g0;
        g0[0] = 1u;                                        // count=1, user D#
        g0[1] = lds_off;                                   // lds_addr
        g0[2] = (unsigned)ga;                              // global_addr[31:0]
        g0[3] = (unsigned)((ga >> 32) & 0x01FFFFFFull)     // global_addr[56:32]
              | 0x80000000u;                               // type=2 ("image")

        i32x8 g1;
        g1[0] = 0x00010000;          // workgroup_mask=0, data_size=1 (2 bytes)
        g1[1] = (int)(256u << 16);   // tensor_dim0[15:0]=256 in bits 63:48
        g1[2] = (int)(64u  << 16);   // tensor_dim1[15:0]=64  in bits 95:80
        g1[3] = (int)(256u << 16);   // tile_dim0=256         in bits 127:112
        g1[4] = 64;                  // tile_dim1=64, tile_dim2=0
        g1[5] = 256;                 // tensor_dim0_stride[31:0]=256
        g1[6] = 0;                   // stride hi / dim1_stride lo
        g1[7] = 0;

        i32x4 g2 = {0, 0, 0, 0};     // 2D tensor: groups 2/3 unused
        i32x4 g3 = {0, 0, 0, 0};
#if __clang_major__ >= 23
        i32x8 g4 = {0, 0, 0, 0, 0, 0, 0, 0};
        __builtin_amdgcn_tensor_load_to_lds(g0, g1, g2, g3, g4, 0);
#else
        __builtin_amdgcn_tensor_load_to_lds(g0, g1, g2, g3, 0);
#endif
        __builtin_amdgcn_s_wait_tensorcnt(0);  // tile in LDS before we signal
    }
    __syncthreads();

    const int tilesM = (M + 15) >> 4;
    const int tm = blockIdx.x * 8 + wave;    // wave-uniform
    if (tm < tilesM) {
        const int r   = lane & 15;           // A row in tile / B column in N-tile
        const int hi  = lane >> 4;           // half-wave K split
        int arow  = tm * 16 + r;
        int arowc = arow < M ? arow : (M - 1);
        const float* Arow = A + (size_t)arowc * KDIM;

        v8f acc[4] = {};
        #pragma unroll
        for (int k0 = 0; k0 < KDIM; k0 += 32) {
            if (k0 + 32 < KDIM)
                __builtin_prefetch(Arow + k0 + 32, 0, 3);

            // A fragment: halves 0..7  = K [k0+8*hi,    +8)
            //             halves 8..15 = K [k0+16+8*hi, +8)   (contiguous f32 runs)
            float4 a0 = *(const float4*)(Arow + k0 + hi * 8);
            float4 a1 = *(const float4*)(Arow + k0 + hi * 8 + 4);
            float4 a2 = *(const float4*)(Arow + k0 + 16 + hi * 8);
            float4 a3 = *(const float4*)(Arow + k0 + 16 + hi * 8 + 4);
            v16h af;
            af[0]  = (_Float16)a0.x; af[1]  = (_Float16)a0.y;
            af[2]  = (_Float16)a0.z; af[3]  = (_Float16)a0.w;
            af[4]  = (_Float16)a1.x; af[5]  = (_Float16)a1.y;
            af[6]  = (_Float16)a1.z; af[7]  = (_Float16)a1.w;
            af[8]  = (_Float16)a2.x; af[9]  = (_Float16)a2.y;
            af[10] = (_Float16)a2.z; af[11] = (_Float16)a2.w;
            af[12] = (_Float16)a3.x; af[13] = (_Float16)a3.y;
            af[14] = (_Float16)a3.z; af[15] = (_Float16)a3.w;

            // 4 B fragments from LDS: 2x ds_load_b128 each
            #pragma unroll
            for (int nt = 0; nt < 4; ++nt) {
                const _Float16* bp = smem + (nt * 16 + r) * KDIM;
                V16 bu;
                bu.h[0] = *(const v8h*)(bp + k0 + hi * 8);
                bu.h[1] = *(const v8h*)(bp + k0 + 16 + hi * 8);
                acc[nt] = __builtin_amdgcn_wmma_f32_16x16x32_f16(
                    false, af, false, bu.v, (short)0, acc[nt], false, false);
            }
        }

        // ---- store: C layout VGPR i -> row i + 8*hi, col = lane&15 ---------
        const bool fullTile = (tm * 16 + 16 <= M);   // wave-uniform
        #pragma unroll
        for (int nt = 0; nt < 4; ++nt) {
            int n = nBase + nt * 16 + r;
            float bv = bias ? bias[n] : 0.0f;
            if (fullTile) {
                #pragma unroll
                for (int i = 0; i < 8; ++i)
                    out[(size_t)(tm * 16 + i + hi * 8) * N + n] = acc[nt][i] + bv;
            } else {
                #pragma unroll
                for (int i = 0; i < 8; ++i) {
                    int row = tm * 16 + i + hi * 8;
                    if (row < M) out[(size_t)row * N + n] = acc[nt][i] + bv;
                }
            }
        }
    }
}

// ---------------------------------------------------------------------------
// Softmax over 16 contiguous logits per (b,q,h) row.
// ---------------------------------------------------------------------------
__global__ void softmax16_kernel(const float* __restrict__ logits,
                                 float* __restrict__ outp, int R)
{
    int r = blockIdx.x * blockDim.x + threadIdx.x;
    if (r >= R) return;
    const float* p = logits + (size_t)r * 16;
    float m = -1e30f;
    #pragma unroll
    for (int i = 0; i < 16; ++i) m = fmaxf(m, p[i]);
    float e[16];
    float s = 0.0f;
    #pragma unroll
    for (int i = 0; i < 16; ++i) { e[i] = __expf(p[i] - m); s += e[i]; }
    float inv = 1.0f / s;
    float* o = outp + (size_t)r * 16;
    #pragma unroll
    for (int i = 0; i < 16; ++i) o[i] = e[i] * inv;
}

// ---------------------------------------------------------------------------
// Deformable bilinear sampling. One wave32 per (b,q,h); lane = head-dim
// channel (HEAD_DIM == 32 == wave32). All control flow is lane-uniform.
// vproj (54 MB) is L2-resident (192 MB L2) so gathers hit L2, not HBM.
// ---------------------------------------------------------------------------
__global__ __launch_bounds__(256) void msda_sample_kernel(
    const float* __restrict__ vproj,   // [bs, Lv, H*D]
    const float* __restrict__ refpts,  // [bs, Lq, L, 2]
    const float* __restrict__ off,     // [bs, Lq, H, L, P, 2]
    const float* __restrict__ attn,    // [bs, Lq, H, L*P]
    const int*   __restrict__ shapes,  // [L, 2]  (H_l, W_l)
    const int*   __restrict__ startix, // [L]
    float* __restrict__ hout,          // [bs, Lq, H*D]
    int bs, int Lq, int Lv)
{
    const int wave = threadIdx.x >> 5;
    const int lane = threadIdx.x & 31;
    long gw = (long)blockIdx.x * 8 + wave;   // (b*Lq + q)*H + h
    long total = (long)bs * Lq * HEADS;
    if (gw >= total) return;

    int  h  = (int)(gw % HEADS);
    long bq = gw / HEADS;
    int  b  = (int)(bq / Lq);

    const float* ref   = refpts + (size_t)bq * LEVELS * 2;
    const float* offp  = off    + ((size_t)bq * HEADS + h) * (LEVELS * POINTS * 2);
    const float* ap    = attn   + ((size_t)bq * HEADS + h) * (LEVELS * POINTS);
    const float* vbase = vproj  + (size_t)b * Lv * EMBED + h * HEAD_DIM + lane;

    float acc = 0.0f;
    #pragma unroll
    for (int l = 0; l < LEVELS; ++l) {
        int Hl = shapes[l * 2 + 0];
        int Wl = shapes[l * 2 + 1];
        float fW = (float)Wl, fH = (float)Hl;
        int st = startix[l];
        float rx = ref[l * 2 + 0];
        float ry = ref[l * 2 + 1];
        #pragma unroll
        for (int p = 0; p < POINTS; ++p) {
            int s = l * POINTS + p;
            // loc = ref + off/(W,H); pixel = loc*size - 0.5 (align_corners=False)
            float x = (rx + offp[s * 2 + 0] / fW) * fW - 0.5f;
            float y = (ry + offp[s * 2 + 1] / fH) * fH - 0.5f;
            float x0f = floorf(x), y0f = floorf(y);
            float fx = x - x0f, fy = y - y0f;
            int x0 = (int)x0f, y0 = (int)y0f;
            float aw = ap[s];

            float w00 = (1.0f - fx) * (1.0f - fy);
            float w10 = fx * (1.0f - fy);
            float w01 = (1.0f - fx) * fy;
            float w11 = fx * fy;

            float sum = 0.0f;
            #pragma unroll
            for (int c = 0; c < 4; ++c) {
                int xi = x0 + (c & 1);
                int yi = y0 + (c >> 1);
                float w = (c == 0) ? w00 : (c == 1) ? w10 : (c == 2) ? w01 : w11;
                if (xi >= 0 && xi < Wl && yi >= 0 && yi < Hl) {  // lane-uniform
                    int idx = st + yi * Wl + xi;
                    sum += w * vbase[(size_t)idx * EMBED];
                }
            }
            acc += aw * sum;
        }
    }
    hout[(size_t)bq * EMBED + h * HEAD_DIM + lane] = acc;
}

// ---------------------------------------------------------------------------
extern "C" void kernel_launch(void* const* d_in, const int* in_sizes, int n_in,
                              void* d_out, int out_size, void* d_ws, size_t ws_size,
                              hipStream_t stream)
{
    const float* query   = (const float*)d_in[0];
    const float* refpts  = (const float*)d_in[1];
    const float* value   = (const float*)d_in[2];
    const int*   shapes  = (const int*)d_in[3];
    const int*   startix = (const int*)d_in[4];
    const float* w_value = (const float*)d_in[5];
    const float* b_value = (const float*)d_in[6];
    const float* w_off   = (const float*)d_in[7];
    const float* b_off   = (const float*)d_in[8];
    const float* w_attn  = (const float*)d_in[9];
    const float* b_attn  = (const float*)d_in[10];
    const float* w_out   = (const float*)d_in[11];
    const float* b_out   = (const float*)d_in[12];
    float* out = (float*)d_out;

    const int bs = 4;
    const int Lq = in_sizes[0] / (bs * EMBED);   // 900
    const int Lv = in_sizes[2] / (bs * EMBED);   // 13294
    const int Mq = bs * Lq;                      // 3600
    const int Mv = bs * Lv;                      // 53176
    const int HLP = HEADS * LEVELS * POINTS;     // 128

    // workspace carve-up (floats first, then f16 weight copies)
    float* vproj   = (float*)d_ws;                       // Mv * 256
    float* offbuf  = vproj   + (size_t)Mv * EMBED;       // Mq * 256
    float* attnlog = offbuf  + (size_t)Mq * EMBED;       // Mq * 128
    float* attnsm  = attnlog + (size_t)Mq * HLP;         // Mq * 128
    float* hout    = attnsm  + (size_t)Mq * HLP;         // Mq * 256
    _Float16* wv16 = (_Float16*)(hout + (size_t)Mq * EMBED); // 256*256
    _Float16* wo16 = wv16 + EMBED * KDIM;                    // 256*256
    _Float16* wa16 = wo16 + EMBED * KDIM;                    // 128*256
    _Float16* wu16 = wa16 + HLP * KDIM;                      // 256*256

    // 0) convert the four weight matrices to f16 (tiny, one-shot)
    f32_to_f16_kernel<<<(EMBED * KDIM + 255) / 256, 256, 0, stream>>>(w_value, wv16, EMBED * KDIM);
    f32_to_f16_kernel<<<(EMBED * KDIM + 255) / 256, 256, 0, stream>>>(w_off,   wo16, EMBED * KDIM);
    f32_to_f16_kernel<<<(HLP   * KDIM + 255) / 256, 256, 0, stream>>>(w_attn,  wa16, HLP * KDIM);
    f32_to_f16_kernel<<<(EMBED * KDIM + 255) / 256, 256, 0, stream>>>(w_out,   wu16, EMBED * KDIM);

    // 1) value projection: vproj = value @ w_value^T + b_value
    {
        dim3 grid((((Mv + 15) / 16) + 7) / 8, EMBED / 64);
        wmma_gemm_bias_kernel<<<grid, 256, 0, stream>>>(
            value, wv16, b_value, vproj, Mv, EMBED);
    }
    // 2) sampling offsets: offbuf = query @ w_off^T + b_off
    {
        dim3 grid((((Mq + 15) / 16) + 7) / 8, EMBED / 64);
        wmma_gemm_bias_kernel<<<grid, 256, 0, stream>>>(
            query, wo16, b_off, offbuf, Mq, EMBED);
    }
    // 3) attention logits: attnlog = query @ w_attn^T + b_attn
    {
        dim3 grid((((Mq + 15) / 16) + 7) / 8, HLP / 64);
        wmma_gemm_bias_kernel<<<grid, 256, 0, stream>>>(
            query, wa16, b_attn, attnlog, Mq, HLP);
    }
    // 4) softmax over 16 (L*P) per (b,q,h)
    {
        int R = Mq * HEADS;
        softmax16_kernel<<<(R + 255) / 256, 256, 0, stream>>>(attnlog, attnsm, R);
    }
    // 5) deformable bilinear sampling -> per-head output
    {
        long waves = (long)Mq * HEADS;
        int blocks = (int)((waves + 7) / 8);
        msda_sample_kernel<<<blocks, 256, 0, stream>>>(
            vproj, refpts, offbuf, attnsm, shapes, startix, hout, bs, Lq, Lv);
    }
    // 6) output projection: out = hout @ w_out^T + b_out
    {
        dim3 grid((((Mq + 15) / 16) + 7) / 8, EMBED / 64);
        wmma_gemm_bias_kernel<<<grid, 256, 0, stream>>>(
            hout, wu16, b_out, out, Mq, EMBED);
    }
}
